// DumbRec_12592844112186
// MI455X (gfx1250) — compile-verified, matrix operand order checked by reading, not run on hardware
//
#include <hip/hip_runtime.h>
#include <hip/hip_bf16.h>
#include <math.h>

// Problem constants (from reference setup_inputs)
#define N_ 4
#define T_ 4096
#define C_ 1024
#define H_ 8
#define D_ 64
#define L_ 64
#define NT_ (N_ * T_)      // 16384
#define HD_ (H_ * D_)      // 512
#define NPROJ_ (3 * HD_)   // 1536 : [qw | qr | v]

typedef __bf16 v16bf __attribute__((ext_vector_type(16)));
typedef float  v8f   __attribute__((ext_vector_type(8)));

// ------------------------------------------------------------------
// f32 -> bf16 elementwise convert
// ------------------------------------------------------------------
__global__ void k_f32_to_bf16(const float* __restrict__ src,
                              __hip_bfloat16* __restrict__ dst, int n) {
  int i = blockIdx.x * blockDim.x + threadIdx.x;
  if (i < n) dst[i] = __float2bfloat16(src[i]);
}

// src [rows, cols] f32 -> dst [cols, rows] bf16   (for w_o transpose)
__global__ void k_transpose_to_bf16(const float* __restrict__ src,
                                    __hip_bfloat16* __restrict__ dst,
                                    int rows, int cols) {
  int i = blockIdx.x * blockDim.x + threadIdx.x;
  if (i < rows * cols) {
    int r = i / cols, c = i % cols;
    dst[(size_t)c * rows + r] = __float2bfloat16(src[i]);
  }
}

// ------------------------------------------------------------------
// bf16 WMMA GEMM:  C[M,N] (f32) = A[M,K] (bf16, row-major) * Bt[N,K]^T
// Per-lane fragment loads follow the CDNA5 16-bit A-matrix layout:
//   lanes 0-15  : row M=lane,    K = {k..k+7} then {k+16..k+23}
//   lanes 16-31 : row M=lane-16, K = {k+8..k+15} then {k+24..k+31}
// B uses the symmetric layout with lane = output column, contiguous K.
// ------------------------------------------------------------------
__device__ inline v16bf ld_frag(const __hip_bfloat16* p) {
  union { uint4 q[2]; v16bf v; } u;
  u.q[0] = *(const uint4*)p;        // 8 bf16 : K .. K+7
  u.q[1] = *(const uint4*)(p + 16); // 8 bf16 : K+16 .. K+23
  return u.v;
}

__global__ __launch_bounds__(256)
void k_gemm_bf16(const __hip_bfloat16* __restrict__ A,   // [M, K]
                 const __hip_bfloat16* __restrict__ Bt,  // [N, K]
                 float* __restrict__ Cc,                 // [M, N]
                 int M, int Nn, int K) {
  const int lane = threadIdx.x & 31;
  const int wave = threadIdx.x >> 5;
  const int m0 = blockIdx.y << 4;                    // 16 rows / block
  const int n0 = ((blockIdx.x << 3) + wave) << 6;    // 64 cols / wave
  if (n0 >= Nn || m0 >= M) return;

  const int half = lane >> 4;  // 0: K-low half, 1: K-high half
  const int r    = lane & 15;
  const __hip_bfloat16* ap = A  + (size_t)(m0 + r) * K + (half << 3);
  const __hip_bfloat16* bp = Bt + (size_t)(n0 + r) * K + (half << 3);
  const size_t bstep = (size_t)16 * K;

  v8f c0 = {}, c1 = {}, c2 = {}, c3 = {};
  for (int k = 0; k < K; k += 32) {
    v16bf a  = ld_frag(ap + k);
    v16bf b0 = ld_frag(bp + k);
    v16bf b1 = ld_frag(bp + bstep + k);
    v16bf b2 = ld_frag(bp + 2 * bstep + k);
    v16bf b3 = ld_frag(bp + 3 * bstep + k);
    c0 = __builtin_amdgcn_wmma_f32_16x16x32_bf16(false, a, false, b0, (short)0, c0, false, false);
    c1 = __builtin_amdgcn_wmma_f32_16x16x32_bf16(false, a, false, b1, (short)0, c1, false, false);
    c2 = __builtin_amdgcn_wmma_f32_16x16x32_bf16(false, a, false, b2, (short)0, c2, false, false);
    c3 = __builtin_amdgcn_wmma_f32_16x16x32_bf16(false, a, false, b3, (short)0, c3, false, false);
  }

  // C/D layout: lanes 0-15 -> N=lane, M = vgpr i; lanes 16-31 -> N=lane-16, M = 8+i
  const int cm = m0 + (half << 3);
  const int cn = n0 + r;
  for (int i = 0; i < 8; ++i) {
    float* row = Cc + (size_t)(cm + i) * Nn + cn;
    row[0]  = c0[i];
    row[16] = c1[i];
    row[32] = c2[i];
    row[48] = c3[i];
  }
}

// ------------------------------------------------------------------
// Per-(n,t): scores vs rotating/static key barrel, softmax over l,
// forget gate A = 1 - sum_h aw, write values V = sum_h aw * v,
// and stash read-addressing ar for the read kernel.
// ------------------------------------------------------------------
__global__ __launch_bounds__(256)
void k_address(const float* __restrict__ proj,   // [NT, 1536] = [qw|qr|v]
               const float* __restrict__ kstar,  // [64, 64]
               float* __restrict__ arbuf,        // [NT, 512]  (h*64+l)
               float* __restrict__ Abuf,         // [N, L, T]
               float* __restrict__ Vbuf) {       // [N, L, T, D]
  __shared__ float qw_s[HD_], qr_s[HD_], v_s[HD_];
  __shared__ float ks_s[L_ * D_];
  __shared__ float sw_s[H_ * L_], sr_s[H_ * L_];

  const int nt  = blockIdx.x;
  const int n   = nt >> 12;         // T = 4096
  const int t   = nt & (T_ - 1);
  const int tid = threadIdx.x;
  const float inv = 0.125f;         // 1/sqrt(64)

  const float* prow = proj + (size_t)nt * NPROJ_;
  for (int i = tid; i < HD_; i += 256) {
    qw_s[i] = prow[i];
    qr_s[i] = prow[HD_ + i];
    v_s[i]  = prow[2 * HD_ + i];
  }
  for (int i = tid; i < L_ * D_; i += 256) ks_s[i] = kstar[i];
  __syncthreads();

  for (int p = tid; p < H_ * L_; p += 256) {
    int h = p >> 6, l = p & 63;
    int lr = (l + t) & 63;          // rotating key barrel
    const float* q1 = qw_s + h * 64;
    const float* q2 = qr_s + h * 64;
    const float* kw = ks_s + lr * 64;
    const float* kr = ks_s + l * 64;
    float sw = 0.f, sr = 0.f;
    for (int d = 0; d < 64; ++d) {
      sw = fmaf(q1[d], kw[d], sw);
      sr = fmaf(q2[d], kr[d], sr);
    }
    sw_s[p] = sw * inv;
    sr_s[p] = sr * inv;
  }
  __syncthreads();

  if (tid < 16) {                   // softmax over l, per h, for both score sets
    float* row = ((tid < 8) ? sw_s : sr_s) + (tid & 7) * 64;
    float m = row[0];
    for (int l = 1; l < 64; ++l) m = fmaxf(m, row[l]);
    float sum = 0.f;
    for (int l = 0; l < 64; ++l) { float e = __expf(row[l] - m); row[l] = e; sum += e; }
    float rinv = 1.f / sum;
    for (int l = 0; l < 64; ++l) row[l] *= rinv;
  }
  __syncthreads();

  if (tid < 64) {                   // forget gate
    int l = tid;
    float s = 0.f;
    for (int h = 0; h < 8; ++h) s += sw_s[h * 64 + l];
    Abuf[(size_t)(n * 64 + l) * T_ + t] = 1.f - s;
  }
  for (int p = tid; p < H_ * L_; p += 256)
    arbuf[(size_t)nt * HD_ + p] = sr_s[p];
  for (int e = tid; e < L_ * D_; e += 256) {   // write values
    int l = e >> 6, d = e & 63;
    float acc = 0.f;
    for (int h = 0; h < 8; ++h) acc = fmaf(sw_s[h * 64 + l], v_s[h * 64 + d], acc);
    Vbuf[((size_t)(n * 64 + l) * T_ + t) * 64 + d] = acc;
  }
}

// ------------------------------------------------------------------
// In-place linear recurrence over time: V[t] <- A[t]*V[t-1] + V[t]
// One block per (n,l); 64 lanes cover d.
// ------------------------------------------------------------------
__global__ __launch_bounds__(64)
void k_scan(const float* __restrict__ Abuf, float* __restrict__ Vbuf) {
  const int nl = blockIdx.x;
  const int d  = threadIdx.x;
  const float* arow = Abuf + (size_t)nl * T_;
  float* vrow = Vbuf + (size_t)nl * T_ * 64 + d;
  float y = 0.f;
  for (int t = 0; t < T_; ++t) {
    float a  = arow[t];
    float xv = vrow[(size_t)t * 64];
    y = fmaf(a, y, xv);
    vrow[(size_t)t * 64] = y;
  }
}

// ------------------------------------------------------------------
// Per-(n,t) read: y[h,d] = sum_l ar[h,l] * rec[l,d] ; store bf16
// ------------------------------------------------------------------
__global__ __launch_bounds__(256)
void k_read(const float* __restrict__ arbuf, const float* __restrict__ Vbuf,
            __hip_bfloat16* __restrict__ yb) {
  __shared__ float rec_s[L_ * D_];
  __shared__ float ar_s[H_ * L_];
  const int nt  = blockIdx.x;
  const int n   = nt >> 12;
  const int t   = nt & (T_ - 1);
  const int tid = threadIdx.x;
  for (int e = tid; e < L_ * D_; e += 256) {
    int l = e >> 6, d = e & 63;
    rec_s[e] = Vbuf[((size_t)(n * 64 + l) * T_ + t) * 64 + d];
  }
  for (int e = tid; e < H_ * L_; e += 256) ar_s[e] = arbuf[(size_t)nt * HD_ + e];
  __syncthreads();
  for (int o = tid; o < HD_; o += 256) {
    int h = o >> 6, d = o & 63;
    float acc = 0.f;
    for (int l = 0; l < 64; ++l) acc = fmaf(ar_s[h * 64 + l], rec_s[l * 64 + d], acc);
    yb[(size_t)nt * HD_ + o] = __float2bfloat16(acc);
  }
}

// ------------------------------------------------------------------
extern "C" void kernel_launch(void* const* d_in, const int* in_sizes, int n_in,
                              void* d_out, int out_size, void* d_ws, size_t ws_size,
                              hipStream_t stream) {
  (void)in_sizes; (void)n_in; (void)out_size; (void)ws_size;
  const float* x     = (const float*)d_in[0];
  const float* kstar = (const float*)d_in[1];
  const float* wqw   = (const float*)d_in[2];
  const float* wqr   = (const float*)d_in[3];
  const float* wv    = (const float*)d_in[4];
  const float* wo    = (const float*)d_in[5];
  float* out = (float*)d_out;

  char* ws = (char*)d_ws;
  size_t off = 0;
  auto alloc = [&](size_t bytes) -> void* {
    void* p = ws + off;
    off = (off + bytes + 255) & ~(size_t)255;
    return p;
  };
  __hip_bfloat16* xb   = (__hip_bfloat16*)alloc((size_t)NT_ * C_ * 2);       // 33.5 MB
  __hip_bfloat16* wf   = (__hip_bfloat16*)alloc((size_t)NPROJ_ * C_ * 2);    //  3.1 MB
  __hip_bfloat16* woT  = (__hip_bfloat16*)alloc((size_t)C_ * HD_ * 2);       //  1.0 MB
  float* proj  = (float*)alloc((size_t)NT_ * NPROJ_ * 4);                    // 100.7 MB
  float* arbuf = (float*)alloc((size_t)NT_ * HD_ * 4);                       // 33.5 MB
  float* Abuf  = (float*)alloc((size_t)N_ * L_ * T_ * 4);                    //  4.2 MB
  float* Vbuf  = (float*)alloc((size_t)N_ * L_ * T_ * D_ * 4);               // 268 MB
  __hip_bfloat16* yb = (__hip_bfloat16*)alloc((size_t)NT_ * HD_ * 2);        // 16.8 MB

  // 1) precision converts
  int nx = NT_ * C_;
  k_f32_to_bf16<<<(nx + 255) / 256, 256, 0, stream>>>(x, xb, nx);
  int nw = HD_ * C_;
  k_f32_to_bf16<<<(nw + 255) / 256, 256, 0, stream>>>(wqw, wf, nw);
  k_f32_to_bf16<<<(nw + 255) / 256, 256, 0, stream>>>(wqr, wf + (size_t)HD_ * C_, nw);
  k_f32_to_bf16<<<(nw + 255) / 256, 256, 0, stream>>>(wv,  wf + (size_t)2 * HD_ * C_, nw);
  k_transpose_to_bf16<<<(nw + 255) / 256, 256, 0, stream>>>(wo, woT, HD_, C_);

  // 2) fused qw/qr/v projection: [16384 x 1024] x [1024 x 1536]
  k_gemm_bf16<<<dim3(NPROJ_ / 512, NT_ / 16), 256, 0, stream>>>(
      xb, wf, proj, NT_, NPROJ_, C_);

  // 3) addressing + softmax + forget gate + write values
  k_address<<<NT_, 256, 0, stream>>>(proj, kstar, arbuf, Abuf, Vbuf);

  // 4) recurrence over time (in-place V -> rec_v)
  k_scan<<<N_ * L_, 64, 0, stream>>>(Abuf, Vbuf);

  // 5) read memory lines -> y (bf16)
  k_read<<<NT_, 256, 0, stream>>>(arbuf, Vbuf, yb);

  // 6) output projection: [16384 x 512] x [512 x 1024] -> d_out (f32)
  k_gemm_bf16<<<dim3(C_ / 512, NT_ / 16), 256, 0, stream>>>(
      yb, woT, out, NT_, C_, HD_);
}
